// MultiHeadRelativeSelfAttention_46325517254727
// MI455X (gfx1250) — matrix operand order repaired and checked by reading, not compiled, not run on hardware
//
#include <hip/hip_runtime.h>
#include <hip/hip_bf16.h>

// ---------------------------------------------------------------------------
// Transformer-XL relative multi-head self-attention for MI455X (gfx1250).
//   B=8, S=1024, dim=1024, H=16, Dh=64
// Matrix math via v_wmma_f32_16x16x32_bf16 (wave32 WMMA, f32 accumulate).
// GEMM staging via GLOBAL_LOAD_ASYNC_TO_LDS_B128 (ASYNCcnt); attention q/k
// tiles via the Tensor Data Mover (TENSOR_LOAD_TO_LDS + s_wait_tensorcnt).
// ---------------------------------------------------------------------------

#define S_LEN   1024
#define DIM_    1024
#define NHEAD   16
#define DHEAD   64
#define BATCH_  8
#define QKV_N   3072   // 3*H*Dh

typedef __attribute__((ext_vector_type(16))) __bf16 v16bf;
typedef __attribute__((ext_vector_type(8)))  float  v8f;
typedef __attribute__((ext_vector_type(4)))  unsigned v4u;
typedef __attribute__((ext_vector_type(8)))  int   v8i;
typedef __attribute__((ext_vector_type(4)))  int   v4i;

union FragBF {
  v16bf v;
  uint4 q[2];
  unsigned short u[16];
};

static __device__ __forceinline__ unsigned short f2bf(float f) {
  unsigned u = __builtin_bit_cast(unsigned, f);
  u += 0x7FFFu + ((u >> 16) & 1u);          // round-to-nearest-even
  return (unsigned short)(u >> 16);
}

static __device__ __forceinline__ v8f v8f_zero() {
  v8f z = {0.f, 0.f, 0.f, 0.f, 0.f, 0.f, 0.f, 0.f};
  return z;
}

// Async global->LDS copy, 16 bytes per lane.  VDST VGPR carries the LDS byte
// offset: for a flat shared-memory pointer, addr[31:0] IS the LDS offset
// so a 32-bit truncation is exact.
static __device__ __forceinline__ void async_g2l_b128(const void* g, void* l) {
  unsigned loff = (unsigned)(unsigned long long)l;
  asm volatile("global_load_async_to_lds_b128 %0, %1, off"
               :: "v"(loff), "v"(g)
               : "memory");
}
static __device__ __forceinline__ void wait_async0() {
  asm volatile("s_wait_asynccnt 0x0" ::: "memory");
}

// ---------------------------------------------------------------------------
// Tensor Data Mover: 2D tile load Global->LDS.
//   width_dw   : tile width in dwords  (tile_dim0, multiple of 4 bytes)
//   rows       : tile height           (tile_dim1)
//   avail_rows : rows available in the tensor from the tile start (clip->0)
//   stride_dw  : row stride in dwords  (tensor_dim0_stride)
// D# bit layout per CDNA5 ISA 8.3/8.4; 2D tile => groups 2/3 (and the extra
// descriptor group of the 6-arg clang-23 builtin) all zero.
// ---------------------------------------------------------------------------
static __device__ __forceinline__ void tdm_load_2d(void* lds, const void* gsrc,
                                                   unsigned width_dw, unsigned rows,
                                                   unsigned avail_rows, unsigned stride_dw) {
  unsigned long long ga = (unsigned long long)gsrc;
  v4u g0;
  g0.x = 1u;                                             // count=1, user mode
  g0.y = (unsigned)(unsigned long long)lds;              // lds_addr (bytes)
  g0.z = (unsigned)ga;                                   // global_addr[31:0]
  g0.w = (unsigned)((ga >> 32) & 0x01FFFFFFu) | (2u << 30);  // addr[56:32] | type=2
  v8i g1;
  g1[0] = (int)(2u << 16);                               // data_size = 4 bytes
  g1[1] = (int)(width_dw << 16);                         // tensor_dim0[15:0]
  g1[2] = (int)((width_dw >> 16) | (avail_rows << 16));  // dim0[31:16] | dim1[15:0]
  g1[3] = (int)((avail_rows >> 16) | (width_dw << 16));  // dim1[31:16] | tile_dim0
  g1[4] = (int)rows;                                     // tile_dim1 | tile_dim2=0
  g1[5] = (int)stride_dw;                                // tensor_dim0_stride[31:0]
  g1[6] = 0;                                             // stride[47:32] | dim1_stride lo
  g1[7] = 0;
  v4i z4 = {0, 0, 0, 0};
  v8i z8 = {0, 0, 0, 0, 0, 0, 0, 0};
  __builtin_amdgcn_tensor_load_to_lds(g0, g1, z4, z4, z8, 0);
}
static __device__ __forceinline__ void wait_tensor0() {
  __builtin_amdgcn_s_wait_tensorcnt(0);
}

// ---------------------------------------------------------------------------
// fp32 -> bf16 conversion
// ---------------------------------------------------------------------------
__global__ void cvt_f32_bf16(const float* __restrict__ src,
                             unsigned short* __restrict__ dst, int n) {
  int i = blockIdx.x * blockDim.x + threadIdx.x;
  if (i < n) dst[i] = f2bf(src[i]);
}

// ---------------------------------------------------------------------------
// Transformer-XL sinusoidal positional embedding, written as bf16 [S, DIM]
// pos_seq[s] = S-1-s ; inv_freq[j] = 10000^(-2j/DIM)
// ---------------------------------------------------------------------------
__global__ void pos_emb_kernel(unsigned short* __restrict__ pos) {
  int idx = blockIdx.x * blockDim.x + threadIdx.x;
  if (idx >= S_LEN * DIM_) return;
  int s = idx >> 10;
  int d = idx & (DIM_ - 1);
  int j = (d < DIM_ / 2) ? d : d - DIM_ / 2;
  // ln(10000) = 9.210340372
  float invf = __expf(-((float)(2 * j) * (1.0f / (float)DIM_)) * 9.2103403719761836f);
  float ang  = (float)(S_LEN - 1 - s) * invf;
  pos[idx] = f2bf((d < DIM_ / 2) ? __sinf(ang) : __cosf(ang));
}

// ---------------------------------------------------------------------------
// bf16 GEMM: C[M,N] = A[M,K] @ B[K,N], f32 accumulate via WMMA.
// Block = 256 threads = 8 wave32 (2x4), tile 128x128, K-step 32.
// A tile staged by async global->LDS; B tile transposed through VGPRs with
// paired-K b32 stores.
// writeF32==0 -> bf16 output Cb ; writeF32==1 -> f32 output Cf + residual.
// ---------------------------------------------------------------------------
__global__ __launch_bounds__(256)
void gemm_bf16_wmma(const unsigned short* __restrict__ A,
                    const unsigned short* __restrict__ B,
                    unsigned short* __restrict__ Cb,
                    float* __restrict__ Cf,
                    const float* __restrict__ res,
                    int M, int N, int K, int writeF32)
{
  __shared__ __align__(16) unsigned short As[128 * 32];   // [row][k]
  __shared__ __align__(16) unsigned short Bt[128 * 32];   // [n][k] (transposed)

  const int t    = threadIdx.x;
  const int wave = t >> 5, lane = t & 31;
  const int wm   = wave >> 2, wn = wave & 3;              // 2 x 4 wave grid
  const int lh   = lane >> 4, l16 = lane & 15;
  const int bm   = blockIdx.y * 128, bn = blockIdx.x * 128;

  v8f acc[4][2];
  #pragma unroll
  for (int i = 0; i < 4; ++i)
    #pragma unroll
    for (int j = 0; j < 2; ++j) acc[i][j] = v8f_zero();

  for (int k0 = 0; k0 < K; k0 += 32) {
    // stage A tile [128 x 32]: two async 16B chunks per thread (no VGPR trip)
    {
      int row = t >> 1, c0 = (t & 1) * 16;
      const unsigned short* g = A + (size_t)(bm + row) * K + k0 + c0;
      unsigned short* l = As + row * 32 + c0;
      async_g2l_b128(g, l);
      async_g2l_b128(g + 8, l + 8);
    }
    // stage B tile [32 x 128] transposed into Bt[n][k]:
    // each thread reads two adjacent K rows (16B each) and writes K-pairs as b32.
    {
      int kp = t >> 4;                // k = 2*kp, 0..31
      int n0 = (t & 15) * 8;
      const unsigned short* s0 = B + (size_t)(k0 + 2 * kp) * N + bn + n0;
      uint4 r0v = *(const uint4*)s0;
      uint4 r1v = *(const uint4*)(s0 + N);
      const unsigned short* r0 = (const unsigned short*)&r0v;
      const unsigned short* r1 = (const unsigned short*)&r1v;
      #pragma unroll
      for (int e = 0; e < 8; ++e) {
        unsigned pk = (unsigned)r0[e] | ((unsigned)r1[e] << 16);
        *(unsigned*)(Bt + (n0 + e) * 32 + 2 * kp) = pk;   // 4B aligned (k even)
      }
    }
    if (k0 + 32 < K) {  // -> global_prefetch_b8
      __builtin_prefetch(A + (size_t)(bm + (t >> 1)) * K + k0 + 32, 0, 1);
      __builtin_prefetch(B + (size_t)(k0 + 32 + (t >> 4) * 2) * N + bn, 0, 1);
    }
    wait_async0();
    __syncthreads();

    FragBF bfr[2];
    #pragma unroll
    for (int nt = 0; nt < 2; ++nt) {
      const unsigned short* bb = Bt + (wn * 32 + nt * 16 + l16) * 32 + 16 * lh;
      bfr[nt].q[0] = ((const uint4*)bb)[0];
      bfr[nt].q[1] = ((const uint4*)bb)[1];
    }
    #pragma unroll
    for (int mt = 0; mt < 4; ++mt) {
      FragBF a;
      const unsigned short* ab = As + (wm * 64 + mt * 16 + l16) * 32;
      a.q[0] = *(const uint4*)(ab + 16 * lh);
      a.q[1] = *(const uint4*)(ab + 16 * lh + 8);
      #pragma unroll
      for (int nt = 0; nt < 2; ++nt) {
        acc[mt][nt] = __builtin_amdgcn_wmma_f32_16x16x32_bf16(
            false, a.v, false, bfr[nt].v, (short)0, acc[mt][nt], false, false);
      }
    }
    __syncthreads();
  }

  // epilogue: C layout row = r + 8*(lane/16), col = lane%16
  #pragma unroll
  for (int mt = 0; mt < 4; ++mt) {
    #pragma unroll
    for (int nt = 0; nt < 2; ++nt) {
      int row0 = bm + wm * 64 + mt * 16 + 8 * lh;
      int col  = bn + wn * 32 + nt * 16 + l16;
      #pragma unroll
      for (int r = 0; r < 8; ++r) {
        size_t off = (size_t)(row0 + r) * N + col;
        float v = acc[mt][nt][r];
        if (writeF32) Cf[off] = v + res[off];
        else          Cb[off] = f2bf(v);
      }
    }
  }
}

// ---------------------------------------------------------------------------
// Fused relative attention, one (b, h, 64-row block) per workgroup.
// 4 wave32s.  Phase 1: U[80x1024] = q @ r^T  (f16 in LDS, 160 KB — only
// possible thanks to CDNA5's 320 KB LDS).  Phase 2: flash-style online
// softmax over 64-wide key tiles; shifted BD term gathered from U.
// q and k tiles arrive via the Tensor Data Mover (one descriptor per tile).
// ---------------------------------------------------------------------------
__global__ __launch_bounds__(128)
void relattn_kernel(const unsigned short* __restrict__ qkv,   // [B*S, 3072] bf16
                    const unsigned short* __restrict__ rbuf,  // [S, 1024]   bf16
                    const unsigned char*  __restrict__ mask,  // [B, S] bool
                    unsigned short*       __restrict__ attn)  // [B*S, 1024] bf16
{
  extern __shared__ __align__(16) char smem[];
  _Float16*       U   = (_Float16*)smem;                         // [80][1024]
  unsigned short* q_s = (unsigned short*)(smem + 80 * 1024 * 2); // [80][64]
  unsigned short* ks  = q_s + 80 * 64;                           // [64][64]  k tile
  unsigned short* vT  = ks + 64 * 64;                            // [64][64]  v^T tile
  unsigned short* pb  = vT + 64 * 64;                            // [64][64]  probs
  float*          msk = (float*)(pb + 64 * 64);                  // [64]

  const int t    = threadIdx.x;
  const int wave = t >> 5, lane = t & 31;
  const int lh   = lane >> 4, l16 = lane & 15;
  const int i0   = blockIdx.x * 64;
  const int h    = blockIdx.y;
  const int b    = blockIdx.z;
  const float scale = 0.125f;   // 1/sqrt(64)

  // ---- stage q rows [i0, i0+80) via TDM (rows past S read as zeros) ----
  if (wave == 0) {
    unsigned avail = (unsigned)(S_LEN - i0);      // rows available from tile start
    tdm_load_2d(q_s,
                qkv + (size_t)(b * S_LEN + i0) * QKV_N + h * DHEAD,
                /*width_dw=*/DHEAD / 2, /*rows=*/80,
                /*avail_rows=*/avail, /*stride_dw=*/QKV_N / 2);
    wait_tensor0();
  }
  __syncthreads();

  // ---- phase 1: U[ii][m] = q[i0+ii] . r[m]  (WMMA, f16 result to LDS) ----
  for (int mt = wave; mt < 5; mt += 4) {       // wave0 also does tile 4 (rows 64..79)
    for (int nt = 0; nt < 64; ++nt) {
      v8f uacc = v8f_zero();
      #pragma unroll
      for (int kk = 0; kk < 2; ++kk) {
        FragBF a, bb;
        const unsigned short* ab = q_s + (mt * 16 + l16) * 64 + kk * 32;
        a.q[0] = *(const uint4*)(ab + 16 * lh);
        a.q[1] = *(const uint4*)(ab + 16 * lh + 8);
        const unsigned short* bbp =
            rbuf + (size_t)(nt * 16 + l16) * 1024 + h * DHEAD + kk * 32 + 16 * lh;
        bb.q[0] = ((const uint4*)bbp)[0];
        bb.q[1] = ((const uint4*)bbp)[1];
        uacc = __builtin_amdgcn_wmma_f32_16x16x32_bf16(
            false, a.v, false, bb.v, (short)0, uacc, false, false);
      }
      _Float16* ub = U + (size_t)(mt * 16 + 8 * lh) * 1024 + nt * 16 + l16;
      #pragma unroll
      for (int r = 0; r < 8; ++r) ub[(size_t)r * 1024] = (_Float16)uacc[r];
    }
  }
  __syncthreads();

  // ---- phase 2: flash attention over 16 key tiles of 64 ----
  float m_run[8], l_run[8];
  v8f o[4];
  #pragma unroll
  for (int r = 0; r < 8; ++r) { m_run[r] = -3.0e38f; l_run[r] = 0.0f; }
  #pragma unroll
  for (int dn = 0; dn < 4; ++dn) o[dn] = v8f_zero();

  for (int jt = 0; jt < 16; ++jt) {
    const int j0 = jt * 64;
    if (wave == 0) {                           // k tile via TDM descriptor
      tdm_load_2d(ks,
                  qkv + (size_t)(b * S_LEN + j0) * QKV_N + 1024 + h * DHEAD,
                  /*width_dw=*/DHEAD / 2, /*rows=*/64,
                  /*avail_rows=*/64, /*stride_dw=*/QKV_N / 2);
    }
    for (int e = t; e < 64 * 64; e += 128) {   // v tile, transposed store
      int jj = e >> 6, d = e & 63;
      vT[d * 64 + jj] = qkv[(size_t)(b * S_LEN + j0 + jj) * QKV_N + 2048 + h * DHEAD + d];
    }
    if (t < 64) msk[t] = mask[(size_t)b * S_LEN + j0 + t] ? 1.0f : 0.0f;
    if (wave == 0) wait_tensor0();
    __syncthreads();

    // AC = q . k^T  for this wave's 16 rows x 64 cols, + shifted BD, + mask
    float sc[4][8];
    #pragma unroll
    for (int jn = 0; jn < 4; ++jn) {
      v8f acc = v8f_zero();
      #pragma unroll
      for (int kk = 0; kk < 2; ++kk) {
        FragBF a, bb;
        const unsigned short* ab = q_s + (wave * 16 + l16) * 64 + kk * 32;
        a.q[0] = *(const uint4*)(ab + 16 * lh);
        a.q[1] = *(const uint4*)(ab + 16 * lh + 8);
        const unsigned short* bbp = ks + (jn * 16 + l16) * 64 + kk * 32 + 16 * lh;
        bb.q[0] = ((const uint4*)bbp)[0];
        bb.q[1] = ((const uint4*)bbp)[1];
        acc = __builtin_amdgcn_wmma_f32_16x16x32_bf16(
            false, a.v, false, bb.v, (short)0, acc, false, false);
      }
      const int j = j0 + jn * 16 + l16;
      const float mval = msk[jn * 16 + l16];
      #pragma unroll
      for (int r = 0; r < 8; ++r) {
        const int iloc = wave * 16 + r + 8 * lh;
        const int irow = i0 + iloc;
        float u;
        if (j <= irow)          u = (float)U[(size_t)iloc * 1024 + (j - irow + S_LEN - 1)];
        else if (j == irow + 1) u = 0.0f;                       // rel-shift zero pad
        else                    u = (float)U[(size_t)(iloc + 1) * 1024 + (j - irow - 2)];
        float s = (acc[r] + u) * scale;
        sc[jn][r] = s * mval - 1.0e30f * (1.0f - mval);
      }
    }

    // online softmax: each row lives across 16 lanes of one half-wave
    #pragma unroll
    for (int r = 0; r < 8; ++r) {
      float mx = fmaxf(fmaxf(sc[0][r], sc[1][r]), fmaxf(sc[2][r], sc[3][r]));
      #pragma unroll
      for (int d = 8; d >= 1; d >>= 1) mx = fmaxf(mx, __shfl_xor(mx, d, 32));
      float mnew = fmaxf(m_run[r], mx);
      float corr = __expf(m_run[r] - mnew);
      float psum = 0.0f;
      #pragma unroll
      for (int jn = 0; jn < 4; ++jn) {
        float p = __expf(sc[jn][r] - mnew);
        sc[jn][r] = p;
        psum += p;
      }
      #pragma unroll
      for (int d = 8; d >= 1; d >>= 1) psum += __shfl_xor(psum, d, 32);
      l_run[r] = l_run[r] * corr + psum;
      m_run[r] = mnew;
      #pragma unroll
      for (int dn = 0; dn < 4; ++dn) o[dn][r] *= corr;
    }

    // stash probabilities (C layout -> LDS -> A layout), same-wave in-order LDS
    #pragma unroll
    for (int jn = 0; jn < 4; ++jn)
      #pragma unroll
      for (int r = 0; r < 8; ++r)
        pb[(wave * 16 + r + 8 * lh) * 64 + jn * 16 + l16] = f2bf(sc[jn][r]);

    // O += P . V
    #pragma unroll
    for (int dn = 0; dn < 4; ++dn) {
      #pragma unroll
      for (int kk = 0; kk < 2; ++kk) {
        FragBF a, bb;
        const unsigned short* ab = pb + (wave * 16 + l16) * 64 + kk * 32;
        a.q[0] = *(const uint4*)(ab + 16 * lh);
        a.q[1] = *(const uint4*)(ab + 16 * lh + 8);
        const unsigned short* bbp = vT + (dn * 16 + l16) * 64 + kk * 32 + 16 * lh;
        bb.q[0] = ((const uint4*)bbp)[0];
        bb.q[1] = ((const uint4*)bbp)[1];
        o[dn] = __builtin_amdgcn_wmma_f32_16x16x32_bf16(
            false, a.v, false, bb.v, (short)0, o[dn], false, false);
      }
    }
    __syncthreads();
  }

  // normalize + store attn_vec (bf16)
  #pragma unroll
  for (int dn = 0; dn < 4; ++dn) {
    #pragma unroll
    for (int r = 0; r < 8; ++r) {
      int irow = i0 + wave * 16 + r + 8 * lh;
      int dh   = dn * 16 + l16;
      float v  = o[dn][r] / l_run[r];
      attn[(size_t)(b * S_LEN + irow) * 1024 + h * DHEAD + dh] = f2bf(v);
    }
  }
}

// ---------------------------------------------------------------------------
// Host-side orchestration
// ---------------------------------------------------------------------------
extern "C" void kernel_launch(void* const* d_in, const int* in_sizes, int n_in,
                              void* d_out, int out_size, void* d_ws, size_t ws_size,
                              hipStream_t stream) {
  const float*         inputs = (const float*)d_in[0];
  const unsigned char* mask   = (const unsigned char*)d_in[1];
  const float*         Wqkv   = (const float*)d_in[2];
  const float*         Wr     = (const float*)d_in[3];
  const float*         Wo     = (const float*)d_in[4];
  float*               out    = (float*)d_out;

  const size_t M   = (size_t)BATCH_ * S_LEN;       // 8192
  char*  ws  = (char*)d_ws;
  size_t off = 0;
  auto carve = [&](size_t bytes) {
    char* p = ws + off;
    off += (bytes + 255) & ~(size_t)255;
    return p;
  };
  unsigned short* x_bf    = (unsigned short*)carve(M * DIM_ * 2);
  unsigned short* wqkv_bf = (unsigned short*)carve((size_t)DIM_ * QKV_N * 2);
  unsigned short* wr_bf   = (unsigned short*)carve((size_t)DIM_ * DIM_ * 2);
  unsigned short* wo_bf   = (unsigned short*)carve((size_t)DIM_ * DIM_ * 2);
  unsigned short* pos_bf  = (unsigned short*)carve((size_t)S_LEN * DIM_ * 2);
  unsigned short* qkv_bf  = (unsigned short*)carve(M * QKV_N * 2);
  unsigned short* r_bf    = (unsigned short*)carve((size_t)S_LEN * DIM_ * 2);
  unsigned short* attn_bf = (unsigned short*)carve(M * DIM_ * 2);
  (void)ws_size; (void)in_sizes; (void)n_in; (void)out_size;

  // 1) fp32 -> bf16 conversions
  {
    int n;
    n = (int)(M * DIM_);
    cvt_f32_bf16<<<(n + 255) / 256, 256, 0, stream>>>(inputs, x_bf, n);
    n = DIM_ * QKV_N;
    cvt_f32_bf16<<<(n + 255) / 256, 256, 0, stream>>>(Wqkv, wqkv_bf, n);
    n = DIM_ * DIM_;
    cvt_f32_bf16<<<(n + 255) / 256, 256, 0, stream>>>(Wr, wr_bf, n);
    cvt_f32_bf16<<<(n + 255) / 256, 256, 0, stream>>>(Wo, wo_bf, n);
  }

  // 2) positional embedding (bf16)
  pos_emb_kernel<<<(S_LEN * DIM_) / 256, 256, 0, stream>>>(pos_bf);

  // 3) qkv = x @ Wqkv   [8192 x 3072]
  gemm_bf16_wmma<<<dim3(QKV_N / 128, (int)(M / 128)), 256, 0, stream>>>(
      x_bf, wqkv_bf, qkv_bf, nullptr, nullptr, (int)M, QKV_N, DIM_, 0);

  // 4) r = pos @ Wr     [1024 x 1024]
  gemm_bf16_wmma<<<dim3(DIM_ / 128, S_LEN / 128), 256, 0, stream>>>(
      pos_bf, wr_bf, r_bf, nullptr, nullptr, S_LEN, DIM_, DIM_, 0);

  // 5) fused relative attention (flash-style, rel-shift via U in LDS)
  {
    size_t smem = (size_t)80 * 1024 * 2   // U (f16)
                + (size_t)80 * 64 * 2     // q tile
                + (size_t)64 * 64 * 2 * 3 // k, v^T, prob tiles
                + 64 * 4;                 // mask
    relattn_kernel<<<dim3(S_LEN / 64, NHEAD, BATCH_), 128, smem, stream>>>(
        qkv_bf, r_bf, mask, attn_bf);
  }

  // 6) out = attn_vec @ Wo + inputs   [8192 x 1024], f32
  gemm_bf16_wmma<<<dim3(DIM_ / 128, (int)(M / 128)), 256, 0, stream>>>(
      attn_bf, wo_bf, nullptr, out, inputs, (int)M, DIM_, DIM_, 1);
}
